// STGCN_3530463118112
// MI455X (gfx1250) — matrix operand order repaired and checked by reading, not compiled, not run on hardware
//
#include <hip/hip_runtime.h>

#define B_ 128
#define T_ 512
#define J_ 21
#define C_ 3
#define H_ 64
#define TT 16
#define M1 (TT * J_)   // 336 rows = 21 tiles of 16
#define JH (J_ * H_)   // 1344
#define G3 (3 * H_)    // 192
#define XTT 64         // xproj rows (t) per block

typedef __attribute__((ext_vector_type(16))) _Float16 v16h;
typedef __attribute__((ext_vector_type(8)))  float    v8f;

union H8 { uint4 u; _Float16 f[8]; };

// ---- WMMA fragment helpers (gfx1250 wave32; rows/cols per CDNA5 ISA 7.12.2, K contiguous in A and B) ----
// A (16x32 f16, row-major source, stride elems): lane r=L&15 is row M,
// halves h -> K = 16*hi + h (contiguous, matches ldB)
__device__ __forceinline__ v16h ldA(const _Float16* p, int stride, int lane) {
  int r = lane & 15, hi = lane >> 4;
  const _Float16* q = p + r * stride;
  v16h a;
#pragma unroll
  for (int h = 0; h < 8; ++h) a[h] = q[hi * 16 + h];
#pragma unroll
  for (int h = 0; h < 8; ++h) a[8 + h] = q[hi * 16 + 8 + h];
  return a;
}
// B (32x16 f16) loaded from weight matrix stored row-major as (N rows x K):
// lane n=L&15 is column N, halves h -> K = 16*hi + h (contiguous 32B per lane)
__device__ __forceinline__ v16h ldB(const _Float16* p, int stride, int lane) {
  int n = lane & 15, hi = lane >> 4;
  const _Float16* q = p + n * stride + hi * 16;
  v16h b;
#pragma unroll
  for (int h = 0; h < 16; ++h) b[h] = q[h];
  return b;
}
__device__ __forceinline__ v8f wmma32(v16h a, v16h b, v8f c) {
  return __builtin_amdgcn_wmma_f32_16x16x32_f16(false, a, false, b, (short)0, c,
                                                false, false);
}

// ---- Kernel 0: convert W_ih / W_hh to f16, zero score accumulator ----
__global__ void init_kernel(const float* __restrict__ Wih,
                            const float* __restrict__ Whh,
                            _Float16* __restrict__ WihH,
                            _Float16* __restrict__ WhhH,
                            float* __restrict__ scores) {
  int i = blockIdx.x * blockDim.x + threadIdx.x;
  if (i < G3 * JH) WihH[i] = (_Float16)Wih[i];
  if (i < G3 * H_) WhhH[i] = (_Float16)Whh[i];
  if (i < B_ * J_) scores[i] = 0.0f;
}

// ---- Kernel 1: fused GCN1 + GCN2 for one (b, 16-timestep) tile ----
__global__ __launch_bounds__(256) void gcn12_kernel(
    const float* __restrict__ x, const float* __restrict__ A,
    const float* __restrict__ W1, const float* __restrict__ b1,
    const float* __restrict__ W2, const float* __restrict__ b2,
    const float* __restrict__ Wa,
    _Float16* __restrict__ h2out, float* __restrict__ scores) {
  extern __shared__ char smem[];
  _Float16* sH1  = (_Float16*)smem;          // 336*64 halves (h1, then h2)
  _Float16* sAg2 = sH1 + M1 * H_;            // 336*64 halves
  _Float16* sW2  = sAg2 + M1 * H_;           // 64*64 halves
  float* sA   = (float*)(sW2 + H_ * H_);     // 441
  float* sAg1 = sA + J_ * J_;                // 336*3
  float* sWa  = sAg1 + M1 * C_;              // 64
  float* sW1  = sWa + H_;                    // 64*3
  float* sb1  = sW1 + H_ * C_;               // 64
  float* sScore = sb1 + H_;                  // 21 (+pad)

  const int tid = threadIdx.x;
  const int b = blockIdx.y;
  const int t0 = blockIdx.x * TT;

  for (int i = tid; i < J_ * J_; i += 256) sA[i] = A[i];
  for (int i = tid; i < H_;      i += 256) { sWa[i] = Wa[i]; sb1[i] = b1[i]; }
  for (int i = tid; i < H_ * C_; i += 256) sW1[i] = W1[i];
  for (int i = tid; i < H_ * H_; i += 256) sW2[i] = (_Float16)W2[i];
  if (tid < J_) sScore[tid] = 0.0f;
  __syncthreads();

  // agg1[t,i,c] = sum_j A[i,j] * x[b,t0+t,j,c]
  const float* xb = x + (size_t)(b * T_ + t0) * J_ * C_;
  for (int idx = tid; idx < M1 * C_; idx += 256) {
    int m = idx / C_, c = idx % C_;
    int t = m / J_, i = m % J_;
    const float* xt = xb + (size_t)t * J_ * C_;
    float acc = 0.f;
#pragma unroll
    for (int j = 0; j < J_; ++j) acc += sA[i * J_ + j] * xt[j * C_ + c];
    sAg1[m * C_ + c] = acc;
  }
  __syncthreads();

  // h1 = relu(agg1 @ W1^T + b1)   (K=3 -> VALU, 8 h per thread, b128 stores)
  for (int idx = tid; idx < M1 * 8; idx += 256) {
    int m = idx >> 3, h0 = (idx & 7) << 3;
    float a0 = sAg1[m * C_], a1 = sAg1[m * C_ + 1], a2 = sAg1[m * C_ + 2];
    H8 V;
#pragma unroll
    for (int e = 0; e < 8; ++e) {
      int h = h0 + e;
      float acc = sb1[h] + a0 * sW1[h * C_] + a1 * sW1[h * C_ + 1] +
                  a2 * sW1[h * C_ + 2];
      V.f[e] = (_Float16)fmaxf(acc, 0.f);
    }
    *(uint4*)&sH1[m * H_ + h0] = V.u;
  }
  __syncthreads();

  // agg2[t,i,h0..h0+7] = sum_j A[i,j] * h1[t,j,h0..h0+7]  (b128 DS reads)
  for (int idx = tid; idx < M1 * 8; idx += 256) {
    int m = idx >> 3, h0 = (idx & 7) << 3;
    int t = m / J_, i = m % J_;
    float acc[8] = {};
#pragma unroll
    for (int j = 0; j < J_; ++j) {
      float aj = sA[i * J_ + j];
      H8 U;
      U.u = *(const uint4*)&sH1[(t * J_ + j) * H_ + h0];
#pragma unroll
      for (int e = 0; e < 8; ++e) acc[e] += aj * (float)U.f[e];
    }
    H8 V;
#pragma unroll
    for (int e = 0; e < 8; ++e) V.f[e] = (_Float16)acc[e];
    *(uint4*)&sAg2[m * H_ + h0] = V.u;
  }
  __syncthreads();

  // h2 = relu(agg2 @ W2^T + b2) via WMMA: M=336(21 tiles) N=64(4) K=64(2x32)
  const int wid = tid >> 5, lane = tid & 31;
  for (int tile = wid; tile < 21 * 4; tile += 8) {
    int tm = tile >> 2, tn = tile & 3;
    v8f acc = {};
#pragma unroll
    for (int k0 = 0; k0 < H_; k0 += 32) {
      v16h a = ldA(sAg2 + (tm * 16) * H_ + k0, H_, lane);
      v16h w = ldB(sW2 + (tn * 16) * H_ + k0, H_, lane);
      acc = wmma32(a, w, acc);
    }
    int n = tn * 16 + (lane & 15);
    int r0 = tm * 16 + ((lane >> 4) << 3);
    float bn = b2[n];
#pragma unroll
    for (int v = 0; v < 8; ++v)
      sH1[(r0 + v) * H_ + n] = (_Float16)fmaxf(acc[v] + bn, 0.f);
  }
  __syncthreads();

  // attention score partials: block-local LDS reduction, then 21 global atomics
  for (int m = tid; m < M1; m += 256) {
    float acc = 0.f;
#pragma unroll
    for (int h8 = 0; h8 < 8; ++h8) {
      H8 U;
      U.u = *(const uint4*)&sH1[m * H_ + h8 * 8];
#pragma unroll
      for (int e = 0; e < 8; ++e) acc += (float)U.f[e] * sWa[h8 * 8 + e];
    }
    atomicAdd(&sScore[m % J_], acc);
  }
  __syncthreads();
  if (tid < J_) atomicAdd(&scores[b * J_ + tid], sScore[tid]);

  // stream h2 tile to global (f16, contiguous, 128b stores)
  _Float16* dst = h2out + (size_t)(b * T_ + t0) * JH;
  const uint4* s4 = (const uint4*)sH1;
  uint4* d4 = (uint4*)dst;
  for (int i = tid; i < (M1 * H_) / 8; i += 256) d4[i] = s4[i];
}

// ---- Kernel 2: softmax over joints (one wave per batch) ----
__global__ __launch_bounds__(32) void attn_kernel(const float* __restrict__ scores,
                                                  const float* __restrict__ ba,
                                                  float* __restrict__ attn,
                                                  float* __restrict__ out_attn) {
  int b = blockIdx.x, lane = threadIdx.x;
  float v = (lane < J_) ? scores[b * J_ + lane] * (1.0f / (float)T_) + ba[0]
                        : -1e30f;
  float m = v;
#pragma unroll
  for (int off = 16; off > 0; off >>= 1) m = fmaxf(m, __shfl_xor(m, off, 32));
  float e = (lane < J_) ? __expf(v - m) : 0.f;
  float s = e;
#pragma unroll
  for (int off = 16; off > 0; off >>= 1) s += __shfl_xor(s, off, 32);
  float a = e / s;
  if (lane < J_) {
    attn[b * J_ + lane] = a;
    out_attn[b * J_ + lane] = a;
  }
}

// ---- Kernel 3: xproj = (attn*h2) @ W_ih^T + b_ih ----
// M=64 rows per block (4 m-tiles), N=192 (12 n-tiles), K=1344.
// 8 waves: wave = (mg in 0..1) x (ng in 0..3); each owns 2 m-tiles x 3 n-tiles.
__global__ __launch_bounds__(256) void xproj_kernel(
    const _Float16* __restrict__ h2, const float* __restrict__ attn,
    const _Float16* __restrict__ WihH, const float* __restrict__ b_ih,
    float* __restrict__ xproj) {
  extern __shared__ char smem[];
  _Float16* sSeq = (_Float16*)smem;            // 64*1344 halves (172KB)
  float* sAttn = (float*)(sSeq + XTT * JH);    // 21 (+pad)

  int tid = threadIdx.x;
  int b = blockIdx.y, t0 = blockIdx.x * XTT;
  if (tid < J_) sAttn[tid] = attn[b * J_ + tid];
  __syncthreads();

  // load + attention-scale the 64x1344 h2 tile (8 halves per b128)
  const uint4* src4 = (const uint4*)(h2 + (size_t)(b * T_ + t0) * JH);
  uint4* seq4 = (uint4*)sSeq;
  for (int i4 = tid; i4 < (XTT * JH) / 8; i4 += 256) {
    H8 U, V;
    U.u = src4[i4];
    float aj = sAttn[((i4 * 8) % JH) >> 6];
#pragma unroll
    for (int e = 0; e < 8; ++e) V.f[e] = (_Float16)((float)U.f[e] * aj);
    seq4[i4] = V.u;
  }
  __syncthreads();

  int wid = tid >> 5, lane = tid & 31;
  int mg = wid & 1, ng = wid >> 1;
  v8f acc[2][3] = {};
  for (int k0 = 0; k0 < JH; k0 += 32) {
    v16h a0 = ldA(sSeq + (mg * 32) * JH + k0, JH, lane);
    v16h a1 = ldA(sSeq + (mg * 32 + 16) * JH + k0, JH, lane);
#pragma unroll
    for (int i = 0; i < 3; ++i) {
      const _Float16* wp = WihH + (size_t)((ng * 3 + i) * 16) * JH + k0;
      if (i == 0) __builtin_prefetch(wp + 32, 0, 1);  // next K chunk of W_ih
      v16h w = ldB(wp, JH, lane);
      acc[0][i] = wmma32(a0, w, acc[0][i]);
      acc[1][i] = wmma32(a1, w, acc[1][i]);
    }
  }
  int hi = lane >> 4, col = lane & 15;
#pragma unroll
  for (int mi = 0; mi < 2; ++mi) {
#pragma unroll
    for (int i = 0; i < 3; ++i) {
      int n = (ng * 3 + i) * 16 + col;
      float bn = b_ih[n];
#pragma unroll
      for (int v = 0; v < 8; ++v) {
        int t = t0 + mg * 32 + mi * 16 + v + 8 * hi;
        xproj[((size_t)t * B_ + b) * G3 + n] = acc[mi][i][v] + bn;  // time-major
      }
    }
  }
}

// ---- Kernel 4: GRU scan, 8 workgroups x 16 batch rows (B-parallel) + head ----
__global__ __launch_bounds__(384) void gru_head_kernel(
    const float* __restrict__ xproj, const _Float16* __restrict__ WhhH,
    const float* __restrict__ b_hh, const float* __restrict__ W_head,
    const float* __restrict__ b_head, float* __restrict__ out) {
  extern __shared__ char smem[];
  _Float16* sWhh = (_Float16*)smem;          // 192*64 halves
  _Float16* sHh  = sWhh + G3 * H_;           // 16*64 halves (hprev f16)
  float* sGh = (float*)(sHh + 16 * H_);      // 16*192 f32
  float* sHf = sGh + 16 * G3;                // 16*64 f32 (hprev f32)

  int tid = threadIdx.x;
  int b0 = blockIdx.x * 16;  // this workgroup's batch rows
  for (int i = tid; i < G3 * H_; i += 384) sWhh[i] = WhhH[i];
  for (int i = tid; i < 16 * H_; i += 384) {
    sHh[i] = (_Float16)0.f;
    sHf[i] = 0.f;
  }
  __syncthreads();

  int wid = tid >> 5, lane = tid & 31;  // 12 waves, one N-tile each
  int col = lane & 15, hi = lane >> 4;
  float bn = b_hh[wid * 16 + col];

  for (int t = 0; t < T_; ++t) {
    // gh = hprev @ Whh^T + b_hh : M=16 (1 tile), N=192 (12 waves), K=64
    v8f acc = {};
#pragma unroll
    for (int k0 = 0; k0 < H_; k0 += 32) {
      v16h a = ldA(sHh + k0, H_, lane);
      v16h w = ldB(sWhh + (wid * 16) * H_ + k0, H_, lane);
      acc = wmma32(a, w, acc);
    }
    {
      int n = wid * 16 + col;
      int r0 = hi << 3;
#pragma unroll
      for (int v = 0; v < 8; ++v) sGh[(r0 + v) * G3 + n] = acc[v] + bn;
    }
    __syncthreads();

    const float* xp = xproj + ((size_t)t * B_ + b0) * G3;
    for (int i = tid; i < 16 * H_; i += 384) {
      int row = i >> 6, h = i & 63;
      float xr = xp[row * G3 + h];
      float xz = xp[row * G3 + 64 + h];
      float xn = xp[row * G3 + 128 + h];
      float hr = sGh[row * G3 + h];
      float hz = sGh[row * G3 + 64 + h];
      float hn = sGh[row * G3 + 128 + h];
      float r = 1.f / (1.f + __expf(-(xr + hr)));
      float z = 1.f / (1.f + __expf(-(xz + hz)));
      float nn = tanhf(xn + r * hn);
      float hv = (1.f - z) * nn + z * sHf[i];
      sHf[i] = hv;
      sHh[i] = (_Float16)hv;
    }
    __syncthreads();
  }

  // out = hT @ W_head^T + b_head for this workgroup's 16 rows
  for (int i = tid; i < 16 * (J_ * C_); i += 384) {
    int row = i / 63, o = i % 63;
    float acc = b_head[o];
#pragma unroll
    for (int h = 0; h < H_; ++h) acc += sHf[row * H_ + h] * W_head[o * H_ + h];
    out[(b0 + row) * 63 + o] = acc;
  }
}

extern "C" void kernel_launch(void* const* d_in, const int* in_sizes, int n_in,
                              void* d_out, int out_size, void* d_ws,
                              size_t ws_size, hipStream_t stream) {
  const float* x      = (const float*)d_in[0];
  const float* A      = (const float*)d_in[1];
  const float* W1     = (const float*)d_in[2];
  const float* b1     = (const float*)d_in[3];
  const float* W2     = (const float*)d_in[4];
  const float* b2     = (const float*)d_in[5];
  const float* Wa     = (const float*)d_in[6];
  const float* ba     = (const float*)d_in[7];
  const float* W_ih   = (const float*)d_in[8];
  const float* b_ih   = (const float*)d_in[9];
  const float* W_hh   = (const float*)d_in[10];
  const float* b_hh   = (const float*)d_in[11];
  const float* W_head = (const float*)d_in[12];
  const float* b_head = (const float*)d_in[13];
  float* out = (float*)d_out;  // [128*63 out | 128*21 attn]

  // workspace layout (256B aligned sections)
  char* ws = (char*)d_ws;
  float*    scores = (float*)ws;                             // 2688 f32
  float*    attn   = (float*)(ws + 10752);                   // 2688 f32
  _Float16* WhhH   = (_Float16*)(ws + 21504);                // 12288 f16
  _Float16* WihH   = (_Float16*)(ws + 46080);                // 258048 f16
  float*    xproj  = (float*)(ws + 562176);                  // 12.6M f32 (time-major)
  _Float16* h2     = (_Float16*)(ws + 562176 + 50331648);    // 88M f16

  init_kernel<<<(G3 * JH + 255) / 256, 256, 0, stream>>>(W_ih, W_hh, WihH,
                                                         WhhH, scores);

  dim3 g1(T_ / TT, B_);
  size_t sm1 = (size_t)(2 * M1 * H_ + H_ * H_) * 2 +
               (size_t)(J_ * J_ + M1 * C_ + H_ + H_ * C_ + H_ + 24) * 4;
  gcn12_kernel<<<g1, 256, sm1, stream>>>(x, A, W1, b1, W2, b2, Wa, h2, scores);

  attn_kernel<<<B_, 32, 0, stream>>>(scores, ba, attn, out + B_ * J_ * C_);

  dim3 g3(T_ / XTT, B_);
  size_t sm2 = (size_t)XTT * JH * 2 + 128;
  xproj_kernel<<<g3, 256, sm2, stream>>>(h2, attn, WihH, b_ih, xproj);

  size_t sm3 = (size_t)G3 * H_ * 2 + (size_t)16 * H_ * 2 +
               (size_t)16 * G3 * 4 + (size_t)16 * H_ * 4;
  gru_head_kernel<<<B_ / 16, 384, sm3, stream>>>(xproj, WhhH, b_hh, W_head,
                                                 b_head, out);
}
